// SpinDistanceEdge_72730976190679
// MI455X (gfx1250) — matrix operand order repaired and checked by reading, not compiled, not run on hardware
//
#include <hip/hip_runtime.h>
#include <stdint.h>

#define N_ATOMS 32768
#define N_EDGES 262144
#define ATOM_F  256
#define EDGE_F  128
#define OUT_F   512
#define N_GAUSS 50
#define IN_F    690
#define K_PAD   704   // 22 k-steps of 32
#define TILE_E  32

typedef __attribute__((ext_vector_type(16))) __bf16        v16bf;
typedef __attribute__((ext_vector_type(8)))  float         v8f;
typedef __attribute__((ext_vector_type(4)))  unsigned int  u32x4;
typedef __attribute__((ext_vector_type(4)))  float         f32x4;
typedef __attribute__((ext_vector_type(4)))  unsigned short u16x4;

static __device__ __forceinline__ unsigned short f2bf(float f) {
    union { float f; uint32_t u; } c; c.f = f;
    uint32_t u = c.u;
    uint32_t r = (u + 0x7FFFu + ((u >> 16) & 1u)) >> 16;  // RNE
    return (unsigned short)r;
}
static __device__ __forceinline__ float bf2f(unsigned short b) {
    union { uint32_t u; float f; } c; c.u = ((uint32_t)b) << 16;
    return c.f;
}
static __device__ __forceinline__ void split4(f32x4 v, u16x4& hi, u16x4& lo) {
    hi.x = f2bf(v.x); lo.x = f2bf(v.x - bf2f(hi.x));
    hi.y = f2bf(v.y); lo.y = f2bf(v.y - bf2f(hi.y));
    hi.z = f2bf(v.z); lo.z = f2bf(v.z - bf2f(hi.z));
    hi.w = f2bf(v.w); lo.w = f2bf(v.w - bf2f(hi.w));
}

// ---------------- W prep: f32 [512][690] -> bf16 hi/lo [512][704] ----------------
__global__ void prep_w_kernel(const float* __restrict__ W,
                              unsigned short* __restrict__ wb_hi,
                              unsigned short* __restrict__ wb_lo) {
    int i = blockIdx.x * blockDim.x + threadIdx.x;
    if (i >= OUT_F * K_PAD) return;
    int r = i / K_PAD, c = i - r * K_PAD;
    float v = (c < IN_F) ? W[r * IN_F + c] : 0.0f;
    unsigned short hi = f2bf(v);
    wb_hi[i] = hi;
    wb_lo[i] = f2bf(v - bf2f(hi));
}

// ---------------- main fused gather + GEMM ----------------
__launch_bounds__(256)
__global__ void edge_gemm_kernel(const float* __restrict__ h,
                                 const float* __restrict__ m,
                                 const float* __restrict__ mag,
                                 const int*   __restrict__ eidx,
                                 const unsigned short* __restrict__ wb_hi,
                                 const unsigned short* __restrict__ wb_lo,
                                 float* __restrict__ out) {
    __shared__ unsigned short sHi[TILE_E * K_PAD];   // 45056 B
    __shared__ unsigned short sLo[TILE_E * K_PAD];   // 45056 B
    __shared__ int   sSrc[TILE_E];
    __shared__ int   sTgt[TILE_E];
    __shared__ float sD[TILE_E];

    const int tid = threadIdx.x;
    const int e0  = blockIdx.x * TILE_E;

    // ---- phase A: indices + spin dot ----
    if (tid < TILE_E) {
        int s = eidx[e0 + tid];
        int t = eidx[N_EDGES + e0 + tid];
        sSrc[tid] = s;
        sTgt[tid] = t;
        float d = mag[s * 3 + 0] * mag[t * 3 + 0]
                + mag[s * 3 + 1] * mag[t * 3 + 1]
                + mag[s * 3 + 2] * mag[t * 3 + 2];
        sD[tid] = d;
    }
    __syncthreads();

    // ---- phase B: build concat feature tile [32][704] as bf16 hi/lo ----
    const f32x4* h4 = (const f32x4*)h;
    const f32x4* m4 = (const f32x4*)m;

    // h[src] -> cols [0,256)
    for (int i = tid; i < TILE_E * (ATOM_F / 4); i += 256) {
        int r = i >> 6, c = i & 63;
        f32x4 v = h4[(long)sSrc[r] * (ATOM_F / 4) + c];
        u16x4 hi, lo; split4(v, hi, lo);
        int off = r * K_PAD + c * 4;
        *(u16x4*)&sHi[off] = hi;
        *(u16x4*)&sLo[off] = lo;
    }
    // h[tgt] -> cols [256,512)
    for (int i = tid; i < TILE_E * (ATOM_F / 4); i += 256) {
        int r = i >> 6, c = i & 63;
        f32x4 v = h4[(long)sTgt[r] * (ATOM_F / 4) + c];
        u16x4 hi, lo; split4(v, hi, lo);
        int off = r * K_PAD + 256 + c * 4;
        *(u16x4*)&sHi[off] = hi;
        *(u16x4*)&sLo[off] = lo;
    }
    // m[e] -> cols [512,640)
    for (int i = tid; i < TILE_E * (EDGE_F / 4); i += 256) {
        int r = i >> 5, c = i & 31;
        f32x4 v = m4[(long)(e0 + r) * (EDGE_F / 4) + c];
        u16x4 hi, lo; split4(v, hi, lo);
        int off = r * K_PAD + 512 + c * 4;
        *(u16x4*)&sHi[off] = hi;
        *(u16x4*)&sLo[off] = lo;
    }
    // Gaussian RBF -> cols [640,690)
    {
        const float dx = 3.0f / 49.0f;
        const float coeff = -0.5f / (dx * dx);
        for (int i = tid; i < TILE_E * N_GAUSS; i += 256) {
            int r = i / N_GAUSS, g = i - r * N_GAUSS;
            float off_g = -1.5f + (float)g * dx;
            float t = sD[r] - off_g;
            float v = __expf(coeff * t * t);
            int off = r * K_PAD + 640 + g;
            unsigned short hi = f2bf(v);
            sHi[off] = hi;
            sLo[off] = f2bf(v - bf2f(hi));
        }
    }
    // zero pad cols [690,704)
    for (int i = tid; i < TILE_E * (K_PAD - IN_F); i += 256) {
        int r = i / (K_PAD - IN_F), c = i - r * (K_PAD - IN_F);
        int off = r * K_PAD + IN_F + c;
        sHi[off] = 0;
        sLo[off] = 0;
    }
    __syncthreads();

    // ---- phase C: WMMA. wave w owns cols [64w, 64w+64), rows [0,32) ----
    const int lane = tid & 31;
    const int wid  = tid >> 5;        // 0..7
    const int l15  = lane & 15;
    const int half = lane >> 4;
    const int nbase = wid * 64;

    v8f acc[2][4];
    #pragma unroll
    for (int mt = 0; mt < 2; ++mt)
        #pragma unroll
        for (int nt = 0; nt < 4; ++nt)
            acc[mt][nt] = {};

    for (int kt = 0; kt < K_PAD / 32; ++kt) {
        const int k0 = kt * 32;

        // A fragments: lane holds row (16*mt + l15), K = k0+8h..+8 and k0+16+8h..+8
        v16bf aHi[2], aLo[2];
        #pragma unroll
        for (int mt = 0; mt < 2; ++mt) {
            int base = (mt * 16 + l15) * K_PAD + k0 + half * 8;
            union { u32x4 q[2]; v16bf v; } uh, ul;
            uh.q[0] = *(const u32x4*)&sHi[base];
            uh.q[1] = *(const u32x4*)&sHi[base + 16];
            ul.q[0] = *(const u32x4*)&sLo[base];
            ul.q[1] = *(const u32x4*)&sLo[base + 16];
            aHi[mt] = uh.v;
            aLo[mt] = ul.v;
        }

        #pragma unroll
        for (int nt = 0; nt < 4; ++nt) {
            // B fragment: lane holds col n, K = k0+8h..+8 and k0+16+8h..+8
            int n = nbase + nt * 16 + l15;
            const unsigned short* bhp = wb_hi + (long)n * K_PAD + k0 + half * 8;
            const unsigned short* blp = wb_lo + (long)n * K_PAD + k0 + half * 8;
            union { u32x4 q[2]; v16bf v; } ubh, ubl;
            ubh.q[0] = *(const u32x4*)bhp;
            ubh.q[1] = *(const u32x4*)(bhp + 16);
            ubl.q[0] = *(const u32x4*)blp;
            ubl.q[1] = *(const u32x4*)(blp + 16);

            #pragma unroll
            for (int mt = 0; mt < 2; ++mt)
                acc[mt][nt] = __builtin_amdgcn_wmma_f32_16x16x32_bf16(
                    false, aHi[mt], false, ubh.v, (short)0, acc[mt][nt], false, false);
            #pragma unroll
            for (int mt = 0; mt < 2; ++mt)
                acc[mt][nt] = __builtin_amdgcn_wmma_f32_16x16x32_bf16(
                    false, aHi[mt], false, ubl.v, (short)0, acc[mt][nt], false, false);
            #pragma unroll
            for (int mt = 0; mt < 2; ++mt)
                acc[mt][nt] = __builtin_amdgcn_wmma_f32_16x16x32_bf16(
                    false, aLo[mt], false, ubh.v, (short)0, acc[mt][nt], false, false);
        }
    }

    // ---- phase D: store. D layout: vgpr r, lane l -> M = r + 8*half, N = l15 ----
    #pragma unroll
    for (int mt = 0; mt < 2; ++mt) {
        #pragma unroll
        for (int nt = 0; nt < 4; ++nt) {
            int col = nbase + nt * 16 + l15;
            #pragma unroll
            for (int r = 0; r < 8; ++r) {
                long row = (long)e0 + mt * 16 + half * 8 + r;
                __builtin_nontemporal_store(acc[mt][nt][r], &out[row * OUT_F + col]);
            }
        }
    }
}

extern "C" void kernel_launch(void* const* d_in, const int* in_sizes, int n_in,
                              void* d_out, int out_size, void* d_ws, size_t ws_size,
                              hipStream_t stream) {
    const float* h   = (const float*)d_in[0];
    const float* m   = (const float*)d_in[1];
    const float* mag = (const float*)d_in[2];
    const int*   ei  = (const int*)d_in[3];
    const float* W   = (const float*)d_in[4];

    unsigned short* wb_hi = (unsigned short*)d_ws;
    unsigned short* wb_lo = wb_hi + (size_t)OUT_F * K_PAD;
    float* out = (float*)d_out;

    int wtot = OUT_F * K_PAD;
    prep_w_kernel<<<(wtot + 255) / 256, 256, 0, stream>>>(W, wb_hi, wb_lo);
    edge_gemm_kernel<<<N_EDGES / TILE_E, 256, 0, stream>>>(h, m, mag, ei, wb_hi, wb_lo, out);
}
